// MixtureOfExperts_77756087927340
// MI455X (gfx1250) — matrix operand order repaired and checked by reading, not compiled, not run on hardware
//
#include <hip/hip_runtime.h>
#include <hip/hip_bf16.h>

// ---------------------------------------------------------------------------
// MoE forward for MI455X (gfx1250, wave32, WMMA f32_16x16x32_f16).
// Roofline: sparse top-2 FLOPs = 68.7 GFLOP; fp32 weight traffic = 268 MB
// -> ~11.5 us HBM floor at 23.3 TB/s. f16 WMMA w/ fp32 accumulate keeps the
// GEMMs near the memory roof; fp32 WMMA (16x16x4) would be compute-bound.
// ---------------------------------------------------------------------------

typedef __attribute__((ext_vector_type(16))) _Float16 v16h;
typedef __attribute__((ext_vector_type(8)))  _Float16 v8h;
typedef __attribute__((ext_vector_type(8)))  float    v8f;

#define NTOK 2048
#define DIM  1024
#define HID  4096
#define NEXP 8
#define MT   32        // tokens per workgroup tile (2 WMMA M-tiles)
#define HCH  128       // hidden chunk per iteration
#define XS   1032      // x LDS row stride in halves (pad 8 -> no bank conflicts)
#define HS   136       // h LDS row stride in halves

// workspace layout (bytes)
static constexpr size_t OFF_CNT   = 0;                                    // 8 * int
static constexpr size_t OFF_LISTS = 64;                                   // 8*2048 uint
static constexpr size_t OFF_GATES = OFF_LISTS + sizeof(unsigned)*NEXP*NTOK;
static constexpr size_t OFF_PROBS = OFF_GATES + sizeof(float)*NTOK*2;
static constexpr size_t OFF_YBUF  = ((OFF_PROBS + sizeof(float)*NTOK*NEXP) + 255) & ~size_t(255);
// ybuf: 2 * NTOK * DIM floats (slot-separated raw expert outputs)

__device__ __forceinline__ float gelu_exact(float v) {
    return 0.5f * v * (1.0f + erff(v * 0.70710678118654752440f));
}

__device__ __forceinline__ v8f wmma_f16(v16h a, v16h b, v8f c) {
    // (neg_a, A, neg_b, B, c_mod, C, reuse_a, reuse_b)
    return __builtin_amdgcn_wmma_f32_16x16x32_f16(false, a, false, b, (short)0, c, false, false);
}

// A-fragment (16x32 f16): lane<16 holds K=[0..15];
// lane>=16 holds K=[16..31] of row M=lane&15 (caller passes row base).
__device__ __forceinline__ v16h load_a_lds(const _Float16* rowBase, int kbase, int lane) {
    int k0 = kbase + ((lane >> 4) << 4);
    v8h lo = *(const v8h*)(rowBase + k0);
    v8h hi = *(const v8h*)(rowBase + k0 + 8);
    return __builtin_shufflevector(lo, hi, 0,1,2,3,4,5,6,7,8,9,10,11,12,13,14,15);
}

// B-fragment (32x16 f16): lane = n + 16*half; lane holds K=[half*16 .. +15] of
// column n. Source is row-major fp32 W with leading dim ld; loads coalesce in
// 64B segments across lanes 0..15 / 16..31, converted f32->f16 in VALU.
__device__ __forceinline__ v16h load_b_global(const float* __restrict__ W, int krow0,
                                              int ld, int colBase, int lane) {
    int kr  = krow0 + ((lane >> 4) << 4);
    const float* p = W + (size_t)kr * ld + colBase + (lane & 15);
    v16h b;
#pragma unroll
    for (int i = 0; i < 16; ++i) b[i] = (_Float16)p[(size_t)i * ld];
    return b;
}

// ---------------------------------------------------------------------------
__global__ void moe_init(int* cnt) {
    if (threadIdx.x < NEXP) cnt[threadIdx.x] = 0;
}

// ---------------------------------------------------------------------------
__global__ __launch_bounds__(256)
void moe_router(const float* __restrict__ x, const float* __restrict__ rw,
                float* __restrict__ probs, float* __restrict__ gates,
                int* __restrict__ cnt, unsigned* __restrict__ lists) {
    int t = blockIdx.x * 256 + threadIdx.x;
    if (t >= NTOK) return;
    const float* xr = x + (size_t)t * DIM;
    float lg[NEXP];
#pragma unroll
    for (int e = 0; e < NEXP; ++e) lg[e] = 0.0f;
    for (int d = 0; d < DIM; d += 4) {
        float4 xv = *(const float4*)(xr + d);
#pragma unroll
        for (int e = 0; e < NEXP; ++e) {
            const float* w = rw + e * DIM + d;
            lg[e] += xv.x * w[0] + xv.y * w[1] + xv.z * w[2] + xv.w * w[3];
        }
    }
    float mx = lg[0];
#pragma unroll
    for (int e = 1; e < NEXP; ++e) mx = fmaxf(mx, lg[e]);
    float p[NEXP], s = 0.0f;
#pragma unroll
    for (int e = 0; e < NEXP; ++e) { p[e] = __expf(lg[e] - mx); s += p[e]; }
    float inv = 1.0f / s;
#pragma unroll
    for (int e = 0; e < NEXP; ++e) { p[e] *= inv; probs[t * NEXP + e] = p[e]; }
    // top-2, jax tie-break = lowest index (strict >)
    int e0 = 0; float p0 = p[0];
#pragma unroll
    for (int e = 1; e < NEXP; ++e) if (p[e] > p0) { p0 = p[e]; e0 = e; }
    int e1 = -1; float p1 = -1.0f;
#pragma unroll
    for (int e = 0; e < NEXP; ++e) if (e != e0 && p[e] > p1) { p1 = p[e]; e1 = e; }
    float gs = 1.0f / (p0 + p1 + 1e-9f);
    gates[t * 2 + 0] = p0 * gs;
    gates[t * 2 + 1] = p1 * gs;
    int pos0 = atomicAdd(&cnt[e0], 1);
    lists[e0 * NTOK + pos0] = (unsigned)t;              // slot 0
    int pos1 = atomicAdd(&cnt[e1], 1);
    lists[e1 * NTOK + pos1] = (unsigned)t | 0x10000u;   // slot 1
}

// ---------------------------------------------------------------------------
// One workgroup = one expert x 32-token tile. 8 waves; each wave owns 128 of
// the 1024 output columns (2 M-tiles x 8 N-tiles of v8f accumulators).
__global__ __launch_bounds__(256, 1)
void moe_expert(const float* __restrict__ x,
                const float* __restrict__ W1, const float* __restrict__ b1,
                const float* __restrict__ W2, const float* __restrict__ b2,
                const int* __restrict__ cnt, const unsigned* __restrict__ lists,
                float* __restrict__ ybuf) {
    const int e    = blockIdx.y;
    const int tile = blockIdx.x;
    const int count = cnt[e];
    const int base  = tile * MT;
    if (base >= count) return;

    const float* W1e = W1 + (size_t)e * DIM * HID;
    const float* W2e = W2 + (size_t)e * HID * DIM;

    __shared__ _Float16 xs[MT * XS];
    __shared__ _Float16 hs[MT * HS];
    __shared__ unsigned toks[MT];

    const int tid  = threadIdx.x;
    const int wave = tid >> 5;
    const int lane = tid & 31;
    const int ln   = lane & 15;

    if (tid < MT) {
        int i = base + tid;
        toks[tid] = (i < count) ? lists[e * NTOK + i] : 0xFFFFFFFFu;
    }
    __syncthreads();

    // Stage 32x1024 fp32 activations -> f16 LDS (float4 coalesced).
#pragma unroll
    for (int v = 0; v < 32; ++v) {
        int fi   = v * 256 + tid;          // float4 index, 256 per row
        int row  = fi >> 8;
        int col4 = fi & 255;
        unsigned ent = toks[row];
        int tok = (ent == 0xFFFFFFFFu) ? 0 : (int)(ent & 0xFFFFu);
        float4 f = ((const float4*)(x + (size_t)tok * DIM))[col4];
        _Float16* dst = &xs[row * XS + col4 * 4];
        dst[0] = (_Float16)f.x; dst[1] = (_Float16)f.y;
        dst[2] = (_Float16)f.z; dst[3] = (_Float16)f.w;
    }
    __syncthreads();

    v8f accY[2][8];
#pragma unroll
    for (int mt = 0; mt < 2; ++mt)
#pragma unroll
        for (int nt = 0; nt < 8; ++nt)
#pragma unroll
            for (int r = 0; r < 8; ++r) accY[mt][nt][r] = 0.0f;

    const _Float16* aRow0 = &xs[ln * XS];
    const _Float16* aRow1 = &xs[(ln + 16) * XS];
    const _Float16* hRow0 = &hs[ln * HS];
    const _Float16* hRow1 = &hs[(ln + 16) * HS];

#pragma unroll 1
    for (int hc = 0; hc < HID / HCH; ++hc) {
        // ---- phase 1: h[32 x 16(this wave)] = gelu(x @ W1 + b1) ----
        const int colh = hc * HCH + wave * 16;   // + ln per lane inside load_b
        float bb = b1[e * HID + colh + ln];
        v8f h0, h1;
#pragma unroll
        for (int r = 0; r < 8; ++r) { h0[r] = bb; h1[r] = bb; }

#pragma unroll 2
        for (int kk = 0; kk < DIM; kk += 32) {
            v16h bfr = load_b_global(W1e, kk, HID, colh, lane);
            v16h a0  = load_a_lds(aRow0, kk, lane);
            v16h a1  = load_a_lds(aRow1, kk, lane);
            h0 = wmma_f16(a0, bfr, h0);
            h1 = wmma_f16(a1, bfr, h1);
        }
        // gelu + deposit into h LDS (C layout: elem r -> row r+8*(lane>>4), col ln)
        const int rhi  = (lane >> 4) << 3;
        const int nloc = wave * 16 + ln;
#pragma unroll
        for (int r = 0; r < 8; ++r) {
            hs[(r + rhi) * HS + nloc]      = (_Float16)gelu_exact(h0[r]);
            hs[(16 + r + rhi) * HS + nloc] = (_Float16)gelu_exact(h1[r]);
        }
        __syncthreads();

        // ---- phase 2: y[32 x 128(this wave)] += h @ W2 ----
#pragma unroll 1
        for (int ks = 0; ks < HCH; ks += 32) {
            v16h a0 = load_a_lds(hRow0, ks, lane);
            v16h a1 = load_a_lds(hRow1, ks, lane);
#pragma unroll
            for (int nt = 0; nt < 8; ++nt) {
                int cold = wave * 128 + nt * 16;
                v16h bfr = load_b_global(W2e, hc * HCH + ks, DIM, cold, lane);
                accY[0][nt] = wmma_f16(a0, bfr, accY[0][nt]);
                accY[1][nt] = wmma_f16(a1, bfr, accY[1][nt]);
            }
        }
        __syncthreads();
    }

    // Epilogue: + b2, scatter to slot buffer (unique writer per (token,slot)).
    const int rhi = (lane >> 4) << 3;
#pragma unroll
    for (int nt = 0; nt < 8; ++nt) {
        int col = wave * 128 + nt * 16 + ln;
        float b2v = b2[e * DIM + col];
#pragma unroll
        for (int mt = 0; mt < 2; ++mt) {
#pragma unroll
            for (int r = 0; r < 8; ++r) {
                int row = mt * 16 + rhi + r;
                if (base + row < count) {
                    unsigned ent = toks[row];
                    int tok  = (int)(ent & 0xFFFFu);
                    int slot = (int)((ent >> 16) & 1u);
                    ybuf[(size_t)slot * NTOK * DIM + (size_t)tok * DIM + col] =
                        accY[mt][nt][r] + b2v;
                }
            }
        }
    }
}

// ---------------------------------------------------------------------------
__global__ __launch_bounds__(256)
void moe_combine(const float* __restrict__ ybuf, const float* __restrict__ gates,
                 float* __restrict__ out) {
    int i = blockIdx.x * 256 + threadIdx.x;       // float4 index over T*D/4
    int t = i >> 8;                               // 256 float4 per token row
    float g0 = gates[t * 2 + 0];
    float g1 = gates[t * 2 + 1];
    float4 y0 = ((const float4*)ybuf)[i];
    float4 y1 = ((const float4*)(ybuf + (size_t)NTOK * DIM))[i];
    float4 o;
    o.x = g0 * y0.x + g1 * y1.x;
    o.y = g0 * y0.y + g1 * y1.y;
    o.z = g0 * y0.z + g1 * y1.z;
    o.w = g0 * y0.w + g1 * y1.w;
    ((float4*)out)[i] = o;
}

// ---------------------------------------------------------------------------
// aux = 0.01 * E * sum_e (count_e/(T*K)) * (sum_t probs[t,e] / T)
// Fixed-order tree reduction -> bit-deterministic.
__global__ __launch_bounds__(256)
void moe_aux(const float* __restrict__ probs, const int* __restrict__ cnt,
             float* __restrict__ out) {
    __shared__ float red[NEXP * 256];
    int tid = threadIdx.x;
    float acc[NEXP];
#pragma unroll
    for (int e = 0; e < NEXP; ++e) acc[e] = 0.0f;
    for (int t = tid; t < NTOK; t += 256) {
#pragma unroll
        for (int e = 0; e < NEXP; ++e) acc[e] += probs[t * NEXP + e];
    }
#pragma unroll
    for (int e = 0; e < NEXP; ++e) red[e * 256 + tid] = acc[e];
    __syncthreads();
    for (int stride = 128; stride > 0; stride >>= 1) {
        if (tid < stride) {
#pragma unroll
            for (int e = 0; e < NEXP; ++e)
                red[e * 256 + tid] += red[e * 256 + tid + stride];
        }
        __syncthreads();
    }
    if (tid == 0) {
        float aux = 0.0f;
#pragma unroll
        for (int e = 0; e < NEXP; ++e) {
            float fe = (float)cnt[e] / (float)(NTOK * 2);
            float pe = red[e * 256] / (float)NTOK;
            aux += fe * pe;
        }
        out[(size_t)NTOK * DIM] = 0.01f * (float)NEXP * aux;
    }
}

// ---------------------------------------------------------------------------
extern "C" void kernel_launch(void* const* d_in, const int* in_sizes, int n_in,
                              void* d_out, int out_size, void* d_ws, size_t ws_size,
                              hipStream_t stream) {
    const float* x   = (const float*)d_in[0];
    const float* rw  = (const float*)d_in[1];
    const float* W1  = (const float*)d_in[2];
    const float* b1  = (const float*)d_in[3];
    const float* W2  = (const float*)d_in[4];
    const float* b2  = (const float*)d_in[5];
    float* out = (float*)d_out;

    char* ws = (char*)d_ws;
    int*      cnt   = (int*)(ws + OFF_CNT);
    unsigned* lists = (unsigned*)(ws + OFF_LISTS);
    float*    gates = (float*)(ws + OFF_GATES);
    float*    probs = (float*)(ws + OFF_PROBS);
    float*    ybuf  = (float*)(ws + OFF_YBUF);

    moe_init<<<1, 32, 0, stream>>>(cnt);
    moe_router<<<NTOK / 256, 256, 0, stream>>>(x, rw, probs, gates, cnt, lists);
    dim3 grid(NTOK / MT, NEXP);
    moe_expert<<<grid, 256, 0, stream>>>(x, W1, b1, W2, b2, cnt, lists, ybuf);
    moe_combine<<<(NTOK * DIM / 4) / 256, 256, 0, stream>>>(ybuf, gates, out);
    moe_aux<<<1, 256, 0, stream>>>(probs, cnt, out);
}
